// DecoderWithAttention_25555055411966
// MI455X (gfx1250) — matrix operand order repaired and checked by reading, not compiled, not run on hardware
//
#include <hip/hip_runtime.h>

// Problem dims (fixed by the reference)
#define B_  256
#define P_  196
#define C_  512
#define D_  512
#define A_  256
#define E_  256
#define V_  193
#define TC_ 256
#define T_  255

typedef __attribute__((ext_vector_type(16))) __bf16 v16bf;
typedef __attribute__((ext_vector_type(8)))  float  v8f;
typedef unsigned int  u32;
typedef unsigned short u16;
typedef __attribute__((ext_vector_type(4))) u32 u32x4;
typedef __attribute__((ext_vector_type(8))) u32 u32x8;

__device__ __forceinline__ u16 f2bf(float f) {
  u32 u = __float_as_uint(f);
  u += 0x7FFFu + ((u >> 16) & 1u);          // round-to-nearest-even
  return (u16)(u >> 16);
}
__device__ __forceinline__ float sigm(float x) { return 1.f / (1.f + __expf(-x)); }

// CDNA5 async global->LDS copy (bypasses VGPRs, tracked by ASYNCcnt).
// lds_addr: LDS byte offset (low 32 bits of a generic LDS pointer).
__device__ __forceinline__ void async_copy_b128(u32 lds_addr, const void* gaddr) {
  asm volatile("global_load_async_to_lds_b128 %0, %1, off"
               :: "v"(lds_addr), "v"((unsigned long long)(uintptr_t)gaddr)
               : "memory");
}
__device__ __forceinline__ void wait_async0() {
  asm volatile("s_wait_asynccnt 0" ::: "memory");
}

// ---------------------------------------------------------------------------
// bf16 WMMA GEMM: Out[m,n] = sum_k X[m,k]*W[n,k] + bias[n]
// X: MxK bf16 row-major (M mult of 64, K mult of 32), W: NxK bf16 row-major.
// WG = 256 threads = 8 waves -> 64(M) x 128(N) tile; wave tile 32x32.
// Double-buffered LDS fed by async copies: 1 barrier / K-step.
// ---------------------------------------------------------------------------
#define XS_STRIDE 40   // 32 + 8 pad (16B aligned rows, conflict-free frags)
#define WS_STRIDE 40

__device__ __forceinline__ v16bf ld_frag(const u16* p0, const u16* p1) {
  u32x4 a0 = *(const u32x4*)p0;
  u32x4 a1 = *(const u32x4*)p1;
  u32x8 cat = {a0.x, a0.y, a0.z, a0.w, a1.x, a1.y, a1.z, a1.w};
  return __builtin_bit_cast(v16bf, cat);
}

__device__ __forceinline__ void gemm_mainloop(
    const u16* __restrict__ X, const u16* __restrict__ W,
    u16 (&Xs)[2][64 * XS_STRIDE], u16 (&Ws)[2][128 * WS_STRIDE],
    int bm, int bn, int N, int K, v8f (&acc)[2][2])
{
  const int tid  = threadIdx.x;
  const int wid  = tid >> 5, lane = tid & 31;
  const int lo16 = lane & 15, hi = lane >> 4;
  const int wm = (wid >> 2) * 32;     // 0 / 32
  const int wn = (wid & 3) * 32;      // 0..96

  // Per-thread staging coordinates
  const int xr = tid >> 2, xc = (tid & 3) * 8;   // X: 8 bf16 (16B) / thread
  const int wr = tid >> 1, wc = (tid & 1) * 16;  // W: 16 bf16 (32B) / thread
  const int nrow = bn + wr;
  // OOB W rows: clamp to row 0 -> garbage only reaches never-stored columns.
  const u16* Xrow = X + (size_t)(bm + xr) * K + xc;
  const u16* Wrow = W + (size_t)(nrow < N ? nrow : 0) * K + wc;

  u32 xdst[2], wdst[2];
#pragma unroll
  for (int q = 0; q < 2; ++q) {
    xdst[q] = (u32)(uintptr_t)&Xs[q][xr * XS_STRIDE + xc];
    wdst[q] = (u32)(uintptr_t)&Ws[q][wr * WS_STRIDE + wc];
  }

  // Prologue: stage tile 0 into buffer 0
  async_copy_b128(xdst[0], Xrow);
  async_copy_b128(wdst[0], Wrow);
  async_copy_b128(wdst[0] + 16, Wrow + 8);
  wait_async0();
  __syncthreads();

  int cur = 0;
  for (int kk = 0; kk < K; kk += 32) {
    const int nk = kk + 32;
    if (nk < K) {                       // stage next tile into other buffer
      const int nb = cur ^ 1;
      async_copy_b128(xdst[nb], Xrow + nk);
      async_copy_b128(wdst[nb], Wrow + nk);
      async_copy_b128(wdst[nb] + 16, Wrow + nk + 8);
      if (nk + 32 < K) __builtin_prefetch(Wrow + nk + 32, 0, 1);
    }

    // Fragments (ISA 16-bit A 16x32 / B 32x16 lane layouts)
    v16bf afrag[2], bfrag[2];
#pragma unroll
    for (int i = 0; i < 2; ++i) {
      const u16* p = &Xs[cur][(wm + i * 16 + lo16) * XS_STRIDE];
      afrag[i] = ld_frag(p + 8 * hi, p + 16 + 8 * hi);
    }
#pragma unroll
    for (int j = 0; j < 2; ++j) {
      const u16* p = &Ws[cur][(wn + j * 16 + lo16) * WS_STRIDE];
      bfrag[j] = ld_frag(p + 8 * hi, p + 16 + 8 * hi);
    }
#pragma unroll
    for (int i = 0; i < 2; ++i)
#pragma unroll
      for (int j = 0; j < 2; ++j)
        acc[i][j] = __builtin_amdgcn_wmma_f32_16x16x32_bf16(
            false, afrag[i], false, bfrag[j], (short)0, acc[i][j], false, false);

    // WMMAs above force s_wait_dscnt on the fragment reads, so every wave's
    // LDS reads of buffer `cur` are complete before it signals this barrier;
    // next iteration's async writes then safely reuse the other buffer.
    wait_async0();
    __syncthreads();
    cur ^= 1;
  }
}

__global__ __launch_bounds__(256) void gemm_bf16_kernel(
    const u16* __restrict__ X, const u16* __restrict__ W,
    const float* __restrict__ bias, float* __restrict__ Out,
    int M, int N, int K)
{
  __shared__ u16 Xs[2][64 * XS_STRIDE];
  __shared__ u16 Ws[2][128 * WS_STRIDE];
  const int tid  = threadIdx.x;
  const int wid  = tid >> 5, lane = tid & 31;
  const int lo16 = lane & 15, hi = lane >> 4;
  const int bm = blockIdx.x * 64;
  const int bn = blockIdx.y * 128;
  const int wm = (wid >> 2) * 32;
  const int wn = (wid & 3) * 32;

  const v8f vzero = {0.f,0.f,0.f,0.f,0.f,0.f,0.f,0.f};
  v8f acc[2][2] = {{vzero, vzero}, {vzero, vzero}};

  gemm_mainloop(X, W, Xs, Ws, bm, bn, N, K, acc);

  // Epilogue: C/D layout: n = lane%16 ; m = r + 8*(lane/16)
#pragma unroll
  for (int j = 0; j < 2; ++j) {
    int n = bn + wn + j * 16 + lo16;
    if (n >= N) continue;
    float bv = bias ? bias[n] : 0.f;
#pragma unroll
    for (int i = 0; i < 2; ++i)
#pragma unroll
      for (int r = 0; r < 8; ++r) {
        int m = bm + wm + i * 16 + r + 8 * hi;
        Out[(size_t)m * N + n] = acc[i][j][r] + bv;
      }
  }
}

// Same GEMM, epilogue writes into predictions[b][t][v] with active masking.
__global__ __launch_bounds__(256) void gemm_preds_kernel(
    const u16* __restrict__ X, const u16* __restrict__ W,
    const float* __restrict__ bias, float* __restrict__ OutPred,
    const int* __restrict__ dlen, int t, int N, int K)
{
  __shared__ u16 Xs[2][64 * XS_STRIDE];
  __shared__ u16 Ws[2][128 * WS_STRIDE];
  const int tid  = threadIdx.x;
  const int wid  = tid >> 5, lane = tid & 31;
  const int lo16 = lane & 15, hi = lane >> 4;
  const int bm = blockIdx.x * 64;
  const int bn = blockIdx.y * 128;
  const int wm = (wid >> 2) * 32;
  const int wn = (wid & 3) * 32;

  const v8f vzero = {0.f,0.f,0.f,0.f,0.f,0.f,0.f,0.f};
  v8f acc[2][2] = {{vzero, vzero}, {vzero, vzero}};

  gemm_mainloop(X, W, Xs, Ws, bm, bn, N, K, acc);

#pragma unroll
  for (int j = 0; j < 2; ++j) {
    int n = bn + wn + j * 16 + lo16;
    if (n >= N) continue;
    float bv = bias[n];
#pragma unroll
    for (int i = 0; i < 2; ++i)
#pragma unroll
      for (int r = 0; r < 8; ++r) {
        int m = bm + wm + i * 16 + r + 8 * hi;   // batch row (sorted order)
        float v = (t < dlen[m]) ? (acc[i][j][r] + bv) : 0.f;
        OutPred[((size_t)m * T_ + t) * V_ + n] = v;
      }
  }
}

// ---------------------------------------------------------------------------
// One-time kernels
// ---------------------------------------------------------------------------

// Stable descending argsort of lens (B=256), caps gather, int/float outputs.
__global__ void sort_kernel(const int* __restrict__ caps,
                            const int* __restrict__ lens2d,
                            int* __restrict__ sidx, int* __restrict__ dlen,
                            int* __restrict__ caps_s,
                            float* __restrict__ out_caps,
                            float* __restrict__ out_dlen,
                            float* __restrict__ out_sidx)
{
  __shared__ int lens[B_];
  __shared__ int sx[B_];
  int tid = threadIdx.x;
  lens[tid] = lens2d[tid];          // caption_lengths is (B,1)
  __syncthreads();
  int l = lens[tid], r = 0;
  for (int j = 0; j < B_; ++j) {
    int lj = lens[j];
    r += (lj > l) || (lj == l && j < tid);
  }
  sx[r] = tid;
  __syncthreads();
  int sb = sx[tid];
  sidx[tid] = sb;
  int dl = lens[sb] - 1;
  dlen[tid] = dl;
  out_dlen[tid] = (float)dl;
  out_sidx[tid] = (float)sb;
  __syncthreads();
  for (int i = tid; i < B_ * TC_; i += B_) {
    int bb = i >> 8, cc = i & (TC_ - 1);
    int v = caps[sx[bb] * TC_ + cc];
    caps_s[i]   = v;
    out_caps[i] = (float)v;
  }
}

__global__ void cvt_kernel(u16* __restrict__ dst, const float* __restrict__ src, int n) {
  int i = blockIdx.x * blockDim.x + threadIdx.x;
  if (i < n) dst[i] = f2bf(src[i]);
}
__global__ void copyf_kernel(float* __restrict__ dst, const float* __restrict__ src, int n) {
  int i = blockIdx.x * blockDim.x + threadIdx.x;
  if (i < n) dst[i] = src[i];
}
__global__ void addf_kernel(float* __restrict__ dst, const float* __restrict__ a,
                            const float* __restrict__ b, int n) {
  int i = blockIdx.x * blockDim.x + threadIdx.x;
  if (i < n) dst[i] = a[i] + b[i];
}

// Wbig[n][k] = k<768 ? W_ih[n][k] : W_hh[n][k-768]   (2048 x 1280, bf16)
__global__ void wbig_kernel(u16* __restrict__ Wbig, const float* __restrict__ W_ih,
                            const float* __restrict__ W_hh) {
  int i = blockIdx.x * blockDim.x + threadIdx.x;
  if (i >= 2048 * 1280) return;
  int n = i / 1280, k = i - n * 1280;
  float v = (k < 768) ? W_ih[(size_t)n * 768 + k] : W_hh[(size_t)n * 512 + (k - 768)];
  Wbig[i] = f2bf(v);
}

// enc_bf[b][p][c] = bf16(enc[sidx[b]][p][c])   (sorted, bf16)
__global__ void encsort_kernel(u16* __restrict__ enc_bf, const float* __restrict__ enc,
                               const int* __restrict__ sidx) {
  int i = blockIdx.x * blockDim.x + threadIdx.x;
  if (i >= B_ * P_ * C_) return;
  int b = i / (P_ * C_), rem = i - b * (P_ * C_);
  enc_bf[i] = f2bf(enc[(size_t)sidx[b] * (P_ * C_) + rem]);
}

// mean over P of sorted enc -> bf16 (B x C)
__global__ void mean_kernel(u16* __restrict__ mean_bf, const float* __restrict__ enc,
                            const int* __restrict__ sidx) {
  int b = blockIdx.x, tid = threadIdx.x;
  int sb = sidx[b];
  for (int c = tid; c < C_; c += blockDim.x) {
    const float* p = enc + (size_t)sb * P_ * C_ + c;
    float acc = 0.f;
    for (int q = 0; q < P_; ++q) acc += p[(size_t)q * C_];
    mean_bf[(size_t)b * C_ + c] = f2bf(acc * (1.f / (float)P_));
  }
}

// hc0 (B x 1024) -> h, c (f32) and h_bf
__global__ void split_hc_kernel(float* __restrict__ h, float* __restrict__ c,
                                u16* __restrict__ h_bf, const float* __restrict__ hc0) {
  int i = blockIdx.x * blockDim.x + threadIdx.x;
  if (i >= B_ * D_) return;
  int b = i >> 9, d = i & 511;
  float hv = hc0[(size_t)b * 1024 + d];
  float cv = hc0[(size_t)b * 1024 + 512 + d];
  h[i] = hv; c[i] = cv; h_bf[i] = f2bf(hv);
}

// ---------------------------------------------------------------------------
// Per-step attention + LSTM input assembly. One WG (256 thr) per batch row.
// enc/enc_att stay f32 and are L2-resident (153 MB < 192 MB).
// ---------------------------------------------------------------------------
__global__ __launch_bounds__(256) void attention_kernel(
    const float* __restrict__ enc,        // original order (B,P,C) f32
    const float* __restrict__ enc_att,    // sorted (B,P,A) f32
    const float* __restrict__ Z,          // (B,768): [dec_att(256)|fbeta(512)]
    const float* __restrict__ wf,         // (A)
    const float* __restrict__ bfull,      // scalar
    const int* __restrict__ sidx, const int* __restrict__ dlen,
    const int* __restrict__ caps_s,       // sorted captions (B,TC) int
    const float* __restrict__ embedding,  // (V,E)
    const u16* __restrict__ h_bf,         // (B,D) bf16
    u16* __restrict__ xcat_bf,            // (B,1280): [emb|gated_awe|h]
    float* __restrict__ alphas_out,       // (B,T,P)
    int t)
{
  __shared__ float dec[A_], wfv[A_], sc[256], red[256];
  int b = blockIdx.x, tid = threadIdx.x;
  dec[tid] = Z[(size_t)b * 768 + tid];
  wfv[tid] = wf[tid];
  __syncthreads();

  // scores[p] = relu(enc_att[b,p,:] + dec_att) . wf + bf
  float s = -1e30f;
  if (tid < P_) {
    const float* ea = enc_att + ((size_t)b * P_ + tid) * A_;
    float acc = bfull[0];
    for (int a = 0; a < A_; ++a) {
      float v = ea[a] + dec[a];
      acc += fmaxf(v, 0.f) * wfv[a];
    }
    s = acc;
  }

  // softmax over p (196)
  red[tid] = s; __syncthreads();
  for (int o = 128; o > 0; o >>= 1) {
    if (tid < o) red[tid] = fmaxf(red[tid], red[tid + o]);
    __syncthreads();
  }
  float mx = red[0]; __syncthreads();
  float e = (tid < P_) ? __expf(s - mx) : 0.f;
  sc[tid] = e;
  red[tid] = e; __syncthreads();
  for (int o = 128; o > 0; o >>= 1) {
    if (tid < o) red[tid] += red[tid + o];
    __syncthreads();
  }
  float inv = 1.f / red[0];

  int active = (t < dlen[b]);
  if (tid < P_)
    alphas_out[((size_t)b * T_ + t) * P_ + tid] = active ? e * inv : 0.f;
  __syncthreads();

  // awe[c] = gate[c] * sum_p alpha[p] * enc[sb,p,c] ; write bf16 into xcat
  int sb = sidx[b];
  for (int c = tid; c < C_; c += 256) {
    const float* ep = enc + (size_t)sb * P_ * C_ + c;
    float acc = 0.f;
    for (int p = 0; p < P_; ++p) acc += sc[p] * ep[(size_t)p * C_];
    acc *= inv;
    float g = sigm(Z[(size_t)b * 768 + 256 + c]);
    xcat_bf[(size_t)b * 1280 + 256 + c] = f2bf(acc * g);
  }
  // emb part
  {
    int tok = caps_s[b * TC_ + t];
    xcat_bf[(size_t)b * 1280 + tid] = f2bf(embedding[(size_t)tok * E_ + tid]);
  }
  // h part (already bf16)
  for (int d = tid; d < D_; d += 256)
    xcat_bf[(size_t)b * 1280 + 768 + d] = h_bf[(size_t)b * D_ + d];
}

// LSTM pointwise: gates (B,2048) f32 -> update h, c, h_bf where active
__global__ void lstm_kernel(const float* __restrict__ gates,
                            float* __restrict__ h, float* __restrict__ c,
                            u16* __restrict__ h_bf,
                            const int* __restrict__ dlen, int t)
{
  int i = blockIdx.x * blockDim.x + threadIdx.x;
  if (i >= B_ * D_) return;
  int b = i >> 9, d = i & 511;
  const float* g = gates + (size_t)b * 2048;
  float ig = sigm(g[d]);
  float fg = sigm(g[512 + d]);
  float gg = tanhf(g[1024 + d]);
  float og = sigm(g[1536 + d]);
  float cn = fg * c[i] + ig * gg;
  float hn = og * tanhf(cn);
  if (t < dlen[b]) { c[i] = cn; h[i] = hn; h_bf[i] = f2bf(hn); }
}

// ---------------------------------------------------------------------------
extern "C" void kernel_launch(void* const* d_in, const int* in_sizes, int n_in,
                              void* d_out, int out_size, void* d_ws, size_t ws_size,
                              hipStream_t stream) {
  const float* enc       = (const float*)d_in[0];
  const int*   caps      = (const int*)  d_in[1];
  const int*   lens      = (const int*)  d_in[2];
  const float* W_enc_att = (const float*)d_in[3];
  const float* b_enc_att = (const float*)d_in[4];
  const float* W_dec_att = (const float*)d_in[5];
  const float* b_dec_att = (const float*)d_in[6];
  const float* W_full    = (const float*)d_in[7];
  const float* b_full    = (const float*)d_in[8];
  const float* embedding = (const float*)d_in[9];
  const float* W_fc      = (const float*)d_in[10];
  const float* b_fc      = (const float*)d_in[11];
  const float* W_ih      = (const float*)d_in[12];
  const float* b_ih      = (const float*)d_in[13];
  const float* W_hh      = (const float*)d_in[14];
  const float* b_hh      = (const float*)d_in[15];
  const float* W_init_h  = (const float*)d_in[16];
  const float* b_init_h  = (const float*)d_in[17];
  const float* W_init_c  = (const float*)d_in[18];
  const float* b_init_c  = (const float*)d_in[19];
  const float* W_fbeta   = (const float*)d_in[20];
  const float* b_fbeta   = (const float*)d_in[21];

  // d_out layout: predictions | caps | decode_lengths | alphas | sort_index
  float* out     = (float*)d_out;
  float* o_pred  = out;
  float* o_caps  = o_pred + (size_t)B_ * T_ * V_;
  float* o_dlen  = o_caps + (size_t)B_ * TC_;
  float* o_alph  = o_dlen + B_;
  float* o_sidx  = o_alph + (size_t)B_ * T_ * P_;

  // Workspace carve-out
  char* ws = (char*)d_ws;
  size_t off = 0;
  auto alloc = [&](size_t bytes) -> char* {
    char* p = ws + off;
    off = (off + bytes + 255) & ~(size_t)255;
    return p;
  };
  int*  sidx    = (int*)alloc(B_ * 4);
  int*  dlen    = (int*)alloc(B_ * 4);
  int*  caps_s  = (int*)alloc((size_t)B_ * TC_ * 4);
  u16*  Wz_bf   = (u16*)alloc((size_t)768 * 512 * 2);      // [W_dec_att; W_fbeta]
  float* bz     = (float*)alloc(768 * 4);
  u16*  Wbig_bf = (u16*)alloc((size_t)2048 * 1280 * 2);    // [W_ih | W_hh]
  float* bbig   = (float*)alloc(2048 * 4);
  u16*  Wfc_bf  = (u16*)alloc((size_t)V_ * D_ * 2);
  u16*  Winit_bf= (u16*)alloc((size_t)1024 * 512 * 2);     // [W_init_h; W_init_c]
  float* binit  = (float*)alloc(1024 * 4);
  u16*  Wenc_bf = (u16*)alloc((size_t)A_ * C_ * 2);
  u16*  enc_bf  = (u16*)alloc((size_t)B_ * P_ * C_ * 2);   // sorted, bf16
  u16*  mean_bf = (u16*)alloc((size_t)B_ * C_ * 2);
  float* enc_att= (float*)alloc((size_t)B_ * P_ * A_ * 4); // sorted, f32
  float* hc0    = (float*)alloc((size_t)B_ * 1024 * 4);
  float* h      = (float*)alloc((size_t)B_ * D_ * 4);
  float* c      = (float*)alloc((size_t)B_ * D_ * 4);
  u16*  h_bf    = (u16*)alloc((size_t)B_ * D_ * 2);
  float* Z      = (float*)alloc((size_t)B_ * 768 * 4);
  u16*  xcat_bf = (u16*)alloc((size_t)B_ * 1280 * 2);
  float* gates  = (float*)alloc((size_t)B_ * 2048 * 4);

  auto cdiv = [](int a, int b) { return (a + b - 1) / b; };

  // --- One-time setup -------------------------------------------------------
  sort_kernel<<<1, 256, 0, stream>>>(caps, lens, sidx, dlen, caps_s,
                                     o_caps, o_dlen, o_sidx);

  cvt_kernel<<<cdiv(256 * 512, 256), 256, 0, stream>>>(Wz_bf, W_dec_att, 256 * 512);
  cvt_kernel<<<cdiv(512 * 512, 256), 256, 0, stream>>>(Wz_bf + 256 * 512, W_fbeta, 512 * 512);
  copyf_kernel<<<1, 256, 0, stream>>>(bz, b_dec_att, 256);
  copyf_kernel<<<2, 256, 0, stream>>>(bz + 256, b_fbeta, 512);

  wbig_kernel<<<cdiv(2048 * 1280, 256), 256, 0, stream>>>(Wbig_bf, W_ih, W_hh);
  addf_kernel<<<cdiv(2048, 256), 256, 0, stream>>>(bbig, b_ih, b_hh, 2048);

  cvt_kernel<<<cdiv(V_ * D_, 256), 256, 0, stream>>>(Wfc_bf, W_fc, V_ * D_);
  cvt_kernel<<<cdiv(512 * 512, 256), 256, 0, stream>>>(Winit_bf, W_init_h, 512 * 512);
  cvt_kernel<<<cdiv(512 * 512, 256), 256, 0, stream>>>(Winit_bf + 512 * 512, W_init_c, 512 * 512);
  copyf_kernel<<<2, 256, 0, stream>>>(binit, b_init_h, 512);
  copyf_kernel<<<2, 256, 0, stream>>>(binit + 512, b_init_c, 512);
  cvt_kernel<<<cdiv(A_ * C_, 256), 256, 0, stream>>>(Wenc_bf, W_enc_att, A_ * C_);

  encsort_kernel<<<cdiv(B_ * P_ * C_, 256), 256, 0, stream>>>(enc_bf, enc, sidx);
  mean_kernel<<<B_, 256, 0, stream>>>(mean_bf, enc, sidx);

  // enc_att = enc_sorted @ W_enc_att^T + b   (M=50176, N=256, K=512)
  gemm_bf16_kernel<<<dim3((B_ * P_) / 64, cdiv(A_, 128)), 256, 0, stream>>>(
      enc_bf, Wenc_bf, b_enc_att, enc_att, B_ * P_, A_, C_);

  // hc0 = mean_enc @ [W_init_h; W_init_c]^T + [b_init_h; b_init_c]
  gemm_bf16_kernel<<<dim3(B_ / 64, 1024 / 128), 256, 0, stream>>>(
      mean_bf, Winit_bf, binit, hc0, B_, 1024, C_);
  split_hc_kernel<<<cdiv(B_ * D_, 256), 256, 0, stream>>>(h, c, h_bf, hc0);

  // --- Sequential decode loop ----------------------------------------------
  for (int t = 0; t < T_; ++t) {
    // Z = h @ [W_dec_att; W_fbeta]^T + bz     (256 x 768, K=512)
    gemm_bf16_kernel<<<dim3(B_ / 64, 768 / 128), 256, 0, stream>>>(
        h_bf, Wz_bf, bz, Z, B_, 768, D_);

    // attention + softmax + gated awe + xcat assembly + alphas output
    attention_kernel<<<B_, 256, 0, stream>>>(
        enc, enc_att, Z, W_full, b_full, sidx, dlen, caps_s, embedding,
        h_bf, xcat_bf, o_alph, t);

    // gates = [emb|awe|h] @ [W_ih|W_hh]^T + (b_ih+b_hh)   (256 x 2048, K=1280)
    gemm_bf16_kernel<<<dim3(B_ / 64, 2048 / 128), 256, 0, stream>>>(
        xcat_bf, Wbig_bf, bbig, gates, B_, 2048, 1280);

    // LSTM cell update (conditional on active)
    lstm_kernel<<<cdiv(B_ * D_, 256), 256, 0, stream>>>(gates, h, c, h_bf, dlen, t);

    // preds = h_new @ W_fc^T + b_fc, masked   (256 x 193, K=512)
    gemm_preds_kernel<<<dim3(B_ / 64, cdiv(V_, 128)), 256, 0, stream>>>(
        h_bf, Wfc_bf, b_fc, o_pred, dlen, t, V_, D_);
  }
}
